// S4DKernel_45543833206900
// MI455X (gfx1250) — matrix operand order repaired and checked by reading, not compiled, not run on hardware
//
#include <hip/hip_runtime.h>

// CDNA5 WMMA vector types (per CDNA5_HIP.md, codegen-verified shapes)
typedef __attribute__((ext_vector_type(16))) _Float16 v16h;
typedef __attribute__((ext_vector_type(8)))  _Float16 v8h;
typedef __attribute__((ext_vector_type(8)))  float    v8f;

#define S4D_H    1024
#define S4D_N2   32
#define S4D_L    4096
#define S4D_L0   64     // outer factor of l
#define S4D_L1   64     // inner factor of l (l = 64*l0 + l1)

// One workgroup (128 threads = 4 waves) per channel h.
// K[h, 64*l0+l1] = Re( U[l0,:] . W[:,l1] ) as a 64x64x64 real GEMM with
// A = [Ur | Ui] (64x64 f16), B = [Wr ; -Wi] (64x64 f16, stored transposed),
// executed with v_wmma_f32_16x16x32_f16 (f32 accumulate).
// Panels are generated with a per-thread complex geometric recurrence:
// each thread owns one mode n and a stride-4 arithmetic sequence of l,
// so only 2 exp + 2 sincos (seed + step) per panel per thread.
__global__ __launch_bounds__(128) void s4d_logvandermonde_wmma(
    const float* __restrict__ log_dt,      // [H]
    const float* __restrict__ Cmat,        // [H, N2]
    const float* __restrict__ Bmat,        // [H, N2]
    const float* __restrict__ log_A_real,  // [H, N2]
    const float* __restrict__ A_imag,      // [H, N2]
    float* __restrict__ out)               // [H, L]
{
    __shared__ _Float16 Uld[S4D_L0][2 * S4D_N2]; // A panel: rows l0, cols k (k<32:Ur, k>=32:Ui)    8 KB
    __shared__ _Float16 Wld[S4D_L1][2 * S4D_N2]; // B^T panel: rows l1, cols k (k<32:Wr, k>=32:-Wi) 8 KB
    __shared__ float pa[S4D_N2], pb[S4D_N2], pcr[S4D_N2], pci[S4D_N2];

    const int h = blockIdx.x;
    const int t = threadIdx.x;

    // ---- Phase 0: per-mode parameters dtA = a + i*w, Ceff = 2*(B*C)*expm1(dtA)/A ----
    if (t < S4D_N2) {
        const int   n  = t;
        const float dt = expf(log_dt[h]);
        const float ar = -expf(log_A_real[h * S4D_N2 + n]);  // Re(A)
        const float ai = A_imag[h * S4D_N2 + n];             // Im(A)
        const float a  = ar * dt;                            // Re(dtA) (<= 0)
        const float w  = ai * dt;                            // Im(dtA)
        // expm1(a + i w) = (expm1(a)*cos w - 2 sin^2(w/2)) + i * exp(a) * sin w
        const float cw = cosf(w);
        const float sw = sinf(w);
        const float sh = sinf(0.5f * w);
        const float er = expm1f(a) * cw - 2.0f * sh * sh;
        const float ei = expf(a) * sw;
        const float bc = Bmat[h * S4D_N2 + n] * Cmat[h * S4D_N2 + n];
        const float inv = 1.0f / (ar * ar + ai * ai);
        // Ceff = bc * expm1(dtA) * conj(A) / |A|^2 ; fold the final factor 2 here
        pcr[n] = 2.0f * bc * (er * ar + ei * ai) * inv;
        pci[n] = 2.0f * bc * (ei * ar - er * ai) * inv;
        pa[n]  = a;
        pb[n]  = w;
    }
    __syncthreads();

    // ---- Phase 1: generate panels via complex geometric recurrences ----
    {
        const int   n  = t & 31;       // mode owned by this thread
        const int   ls = t >> 5;       // starting l (0..3), stride 4
        const float a  = pa[n];
        const float w  = pb[n];

        // ---- U panel: U[l0] = Ceff * exp(dtA*64*l0), l0 = ls, ls+4, ..., ls+60
        {
            float sc, ss;
            // seed: Ceff * exp(dtA * 64 * ls)
            const float e0 = __expf(a * (64.0f * (float)ls));
            __sincosf(w * (64.0f * (float)ls), &ss, &sc);
            float zr = e0 * (pcr[n] * sc - pci[n] * ss);
            float zi = e0 * (pcr[n] * ss + pci[n] * sc);
            // step: exp(dtA * 256)
            const float es = __expf(a * 256.0f);
            float str, sti;
            __sincosf(w * 256.0f, &sti, &str);
            const float stepr = es * str, stepi = es * sti;
#pragma unroll
            for (int k = 0; k < 16; ++k) {
                const int l0 = ls + 4 * k;
                Uld[l0][n]          = (_Float16)zr;
                Uld[l0][n + S4D_N2] = (_Float16)zi;
                const float nzr = zr * stepr - zi * stepi;
                const float nzi = zr * stepi + zi * stepr;
                zr = nzr; zi = nzi;
            }
        }

        // ---- W panel: W[l1] = exp(dtA*l1), l1 = ls, ls+4, ..., ls+60
        {
            float sc, ss;
            const float e0 = __expf(a * (float)ls);
            __sincosf(w * (float)ls, &ss, &sc);
            float zr = e0 * sc;
            float zi = e0 * ss;
            // step: exp(dtA * 4)
            const float es = __expf(a * 4.0f);
            float str, sti;
            __sincosf(w * 4.0f, &sti, &str);
            const float stepr = es * str, stepi = es * sti;
#pragma unroll
            for (int k = 0; k < 16; ++k) {
                const int l1 = ls + 4 * k;
                Wld[l1][n]          = (_Float16)zr;       //  Wr
                Wld[l1][n + S4D_N2] = (_Float16)(-zi);    // -Wi
                const float nzr = zr * stepr - zi * stepi;
                const float nzi = zr * stepi + zi * stepr;
                zr = nzr; zi = nzi;
            }
        }
    }
    __syncthreads();

    // ---- Phase 2: 64x64x64 GEMM with v_wmma_f32_16x16x32_f16 ----
    const int lane = t & 31;
    const int half = lane >> 4;   // 0: lanes 0-15, 1: lanes 16-31
    const int lrow = lane & 15;
    const int mt   = t >> 5;      // wave id -> M-tile row (0..3)

    float* __restrict__ outh = out + (size_t)h * S4D_L;

    for (int nt = 0; nt < 4; ++nt) {
        v8f acc = {};
#pragma unroll
        for (int ks = 0; ks < 2; ++ks) {
            // A fragment: 16-bit A 16x32 layout.
            // lanes 0-15 (half=0): K = ks*32 + {0..7, 16..23}; lanes 16-31: +8.
            const _Float16* ap = &Uld[mt * 16 + lrow][ks * 32 + half * 8];
            v8h a0 = *(const v8h*)(ap);
            v8h a1 = *(const v8h*)(ap + 16);
            v16h af = __builtin_shufflevector(a0, a1, 0, 1, 2, 3, 4, 5, 6, 7,
                                              8, 9, 10, 11, 12, 13, 14, 15);
            // B fragment: column n = nt*16+lrow, same K order as A
            // (lanes 0-15: K = ks*32 + {0..7, 16..23}; lanes 16-31: +8).
            const _Float16* bp = &Wld[nt * 16 + lrow][ks * 32 + half * 8];
            v8h b0 = *(const v8h*)(bp);
            v8h b1 = *(const v8h*)(bp + 16);
            v16h bf = __builtin_shufflevector(b0, b1, 0, 1, 2, 3, 4, 5, 6, 7,
                                              8, 9, 10, 11, 12, 13, 14, 15);
            acc = __builtin_amdgcn_wmma_f32_16x16x32_f16(
                /*neg_a=*/false, af, /*neg_b=*/false, bf,
                /*c_mod=*/(short)0, acc, /*reuse_a=*/false, /*reuse_b=*/false);
        }
        // D layout: VGPR r -> row M = r + 8*half, col N = lane&15
        const int col  = nt * 16 + lrow;
        const int row0 = mt * 16 + half * 8;
#pragma unroll
        for (int r = 0; r < 8; ++r)
            outh[(row0 + r) * S4D_L1 + col] = acc[r];
    }
}

extern "C" void kernel_launch(void* const* d_in, const int* in_sizes, int n_in,
                              void* d_out, int out_size, void* d_ws, size_t ws_size,
                              hipStream_t stream) {
    (void)in_sizes; (void)n_in; (void)out_size; (void)d_ws; (void)ws_size;
    const float* log_dt     = (const float*)d_in[0];
    const float* C          = (const float*)d_in[1];
    const float* B          = (const float*)d_in[2];
    const float* log_A_real = (const float*)d_in[3];
    const float* A_imag     = (const float*)d_in[4];
    // d_in[5] is L (fixed at 4096 in setup_inputs; geometry is compile-time)
    float* out = (float*)d_out;

    s4d_logvandermonde_wmma<<<dim3(S4D_H), dim3(128), 0, stream>>>(
        log_dt, C, B, log_A_real, A_imag, out);
}
